// CrossAttention_30803505447429
// MI455X (gfx1250) — matrix operand order repaired and checked by reading, not compiled, not run on hardware
//
#include <hip/hip_runtime.h>

typedef __attribute__((ext_vector_type(16))) _Float16 v16h;
typedef __attribute__((ext_vector_type(8)))  _Float16 v8h;
typedef __attribute__((ext_vector_type(8)))  float    v8f;

#define N_TOK 9216   // 96*96
#define C_IN  128
#define CQ    16
#define LOG2E 1.4426950408889634f
#define KSPAN (N_TOK / 2)   // keys per split wave

// ---- DPP helpers: allreduce within each 16-lane half of a wave32 ----------
template <int CTRL>
__device__ __forceinline__ float dpp_movf(float x) {
  return __builtin_bit_cast(
      float, __builtin_amdgcn_mov_dpp(__builtin_bit_cast(int, x), CTRL, 0xF, 0xF, true));
}
__device__ __forceinline__ float rowmax16(float x) {
  x = fmaxf(x, dpp_movf<0xB1>(x));    // quad_perm [1,0,3,2]
  x = fmaxf(x, dpp_movf<0x4E>(x));    // quad_perm [2,3,0,1]
  x = fmaxf(x, dpp_movf<0x141>(x));   // row_half_mirror
  x = fmaxf(x, dpp_movf<0x140>(x));   // row_mirror
  return x;
}
__device__ __forceinline__ float rowsum16(float x) {
  x += dpp_movf<0xB1>(x);
  x += dpp_movf<0x4E>(x);
  x += dpp_movf<0x141>(x);
  x += dpp_movf<0x140>(x);
  return x;
}

// ---------------------------------------------------------------------------
// Projection: q (pre-scaled by log2(e)) and kT, both [b][n][16] f16.
// ---------------------------------------------------------------------------
__global__ void qk_proj_kernel(const float* __restrict__ y,
                               const float* __restrict__ wq,
                               const float* __restrict__ wk,
                               _Float16* __restrict__ qf,
                               _Float16* __restrict__ ktf) {
  const int pos = blockIdx.x * blockDim.x + threadIdx.x;
  const int b   = blockIdx.y;
  const float* yb = y + (size_t)b * C_IN * N_TOK + pos;
  float aq[CQ], ak[CQ];
#pragma unroll
  for (int d = 0; d < CQ; ++d) { aq[d] = 0.f; ak[d] = 0.f; }
  for (int c = 0; c < C_IN; ++c) {
    float yv = yb[(size_t)c * N_TOK];
#pragma unroll
    for (int d = 0; d < CQ; ++d) {
      aq[d] = fmaf(wq[d * C_IN + c], yv, aq[d]);
      ak[d] = fmaf(wk[d * C_IN + c], yv, ak[d]);
    }
  }
  _Float16* qo = qf  + ((size_t)b * N_TOK + pos) * CQ;
  _Float16* ko = ktf + ((size_t)b * N_TOK + pos) * CQ;
#pragma unroll
  for (int d = 0; d < CQ; ++d) {
    qo[d] = (_Float16)(aq[d] * LOG2E);   // fold log2e so softmax uses exp2
    ko[d] = (_Float16)ak[d];
  }
}

// ---------------------------------------------------------------------------
// Projection: v[b][cc][pos] f16 = (w_v @ x)[cc][pos]
// ---------------------------------------------------------------------------
__global__ void v_proj_kernel(const float* __restrict__ x,
                              const float* __restrict__ wv,
                              _Float16* __restrict__ vf) {
  const int pos = blockIdx.x * blockDim.x + threadIdx.x;
  const int cc  = blockIdx.y;
  const int b   = blockIdx.z;
  const float* xb   = x  + (size_t)b * C_IN * N_TOK + pos;
  const float* wrow = wv + (size_t)cc * C_IN;
  float acc = 0.f;
#pragma unroll 8
  for (int c = 0; c < C_IN; ++c)
    acc = fmaf(wrow[c], xb[(size_t)c * N_TOK], acc);
  vf[((size_t)b * C_IN + cc) * N_TOK + pos] = (_Float16)acc;
}

// ---------------------------------------------------------------------------
// Flash attention, split-K in-block:
//   block = 8 waves; waves (qslot, split) with split∈{0,1} each cover half
//   the keys for one 16-query tile; partials merged through LDS at the end.
// Per 64-key chunk: 4 QK WMMA -> DPP row-max -> exp2 -> 2x(LDS P-T + 8 PV WMMA)
// ---------------------------------------------------------------------------
__launch_bounds__(256)
__global__ void flash_attn_kernel(const _Float16* __restrict__ qf,
                                  const _Float16* __restrict__ ktf,
                                  const _Float16* __restrict__ vf,
                                  float* __restrict__ out) {
  __shared__ __align__(16) _Float16 plds[8][2][16][32];  // P transpose pads (16 KB)
  __shared__ __align__(32) float    clds[4][8][32][8];   // split1 O tiles   (32 KB)
  __shared__ __align__(16) float    cml[4][2][32][8];    // split1 m,l       ( 8 KB)

  const int lane  = threadIdx.x & 31;
  const int wid   = threadIdx.x >> 5;    // 0..7
  const int qslot = wid & 3;
  const int split = wid >> 2;            // 0 or 1
  const int b     = blockIdx.y;
  const int qbase = (blockIdx.x * 4 + qslot) * 16;

  const int hi = lane >> 4;   // 16-lane half index
  const int lo = lane & 15;

  // ---- Q fragment: A-layout 16x32, real K=0..15, upper half zero -----------
  v16h qa;
  {
    const _Float16* qrow = qf + ((size_t)b * N_TOK + qbase + lo) * CQ + hi * 8;
    v8h qv = *(const v8h*)qrow;
#pragma unroll
    for (int i = 0; i < 8; ++i) { qa[i] = qv[i]; qa[i + 8] = (_Float16)0.f; }
  }

  const _Float16* ktb  = ktf + (size_t)b * N_TOK * CQ;
  const _Float16* vbas = vf  + (size_t)b * C_IN * N_TOK;

  v8f o[8];
#pragma unroll
  for (int f = 0; f < 8; ++f)
#pragma unroll
    for (int r = 0; r < 8; ++r) o[f][r] = 0.f;

  float mrow[8], lpart[8];               // lpart: per-lane partial row sums
#pragma unroll
  for (int r = 0; r < 8; ++r) { mrow[r] = -1e30f; lpart[r] = 0.f; }

  const int kbeg = split * KSPAN;
  const int kend = kbeg + KSPAN;
  for (int kb = kbeg; kb < kend; kb += 64) {
    // ---- K fragments & scores for 4 sixteen-key tiles ---------------------
    v8f s[4];
#pragma unroll
    for (int j = 0; j < 4; ++j) {
      v16h kfr;
#pragma unroll
      for (int i = 0; i < 16; ++i) kfr[i] = (_Float16)0.f;
      {
        v8h kv = *(const v8h*)(ktb + (size_t)(kb + j * 16 + lo) * CQ + hi * 8);
#pragma unroll
        for (int i = 0; i < 8; ++i) kfr[i] = kv[i];
      }
      v8f zacc;
#pragma unroll
      for (int r = 0; r < 8; ++r) zacc[r] = 0.f;
      s[j] = __builtin_amdgcn_wmma_f32_16x16x32_f16(false, qa, false, kfr, (short)0, zacc, false, false);
    }

    // ---- row max over 64 columns (DPP butterflies, no LDS) ----------------
    float sc[8];
    bool noscale = true;
#pragma unroll
    for (int r = 0; r < 8; ++r) {
      float t = fmaxf(fmaxf(s[0][r], s[1][r]), fmaxf(s[2][r], s[3][r]));
      t = rowmax16(t);
      float mn = fmaxf(mrow[r], t);
      sc[r] = __builtin_amdgcn_exp2f(mrow[r] - mn);
      mrow[r] = mn;
      noscale = noscale && (sc[r] == 1.0f);
    }
    if (!__all(noscale)) {               // rescale only while max still moves
#pragma unroll
      for (int r = 0; r < 8; ++r) {
        lpart[r] *= sc[r];
#pragma unroll
        for (int f = 0; f < 8; ++f) o[f][r] *= sc[r];
      }
    }

    // ---- P = exp2(s - m) in place; accumulate lane-partial row sums -------
#pragma unroll
    for (int j = 0; j < 4; ++j)
#pragma unroll
      for (int r = 0; r < 8; ++r)
        s[j][r] = __builtin_amdgcn_exp2f(s[j][r] - mrow[r]);
#pragma unroll
    for (int r = 0; r < 8; ++r)
      lpart[r] += (s[0][r] + s[1][r]) + (s[2][r] + s[3][r]);

    // ---- two 32-key halves: P transpose via LDS, then 8 PV WMMAs ----------
#pragma unroll
    for (int h2 = 0; h2 < 2; ++h2) {
#pragma unroll
      for (int r = 0; r < 8; ++r) {
        plds[wid][h2][r + 8 * hi][lo]      = (_Float16)s[2 * h2][r];
        plds[wid][h2][r + 8 * hi][lo + 16] = (_Float16)s[2 * h2 + 1][r];
      }
      v16h pa;
      {
        v8h plo = *(const v8h*)&plds[wid][h2][lo][hi * 16];
        v8h phi = *(const v8h*)&plds[wid][h2][lo][hi * 16 + 8];
#pragma unroll
        for (int i = 0; i < 8; ++i) { pa[i] = plo[i]; pa[i + 8] = phi[i]; }
      }
      const _Float16* vp = vbas + (size_t)lo * N_TOK + kb + h2 * 32 + hi * 16;
#pragma unroll
      for (int f = 0; f < 8; ++f) {
        v16h vfrag = *(const v16h*)(vp + (size_t)(f * 16) * N_TOK);
        o[f] = __builtin_amdgcn_wmma_f32_16x16x32_f16(false, pa, false, vfrag, (short)0, o[f], false, false);
      }
    }
  }

  // ---- finish split-local sums --------------------------------------------
  float lrow[8];
#pragma unroll
  for (int r = 0; r < 8; ++r) lrow[r] = rowsum16(lpart[r]);

  // ---- split1 publishes its partial state through LDS ---------------------
  if (split == 1) {
#pragma unroll
    for (int f = 0; f < 8; ++f)
      *(v8f*)&clds[qslot][f][lane][0] = o[f];
#pragma unroll
    for (int r = 0; r < 8; ++r) {
      cml[qslot][0][lane][r] = mrow[r];
      cml[qslot][1][lane][r] = lrow[r];
    }
  }
  __syncthreads();

  // ---- split0 merges, normalizes, stores ----------------------------------
  if (split == 0) {
    float a0[8], a1[8], inv[8];
#pragma unroll
    for (int r = 0; r < 8; ++r) {
      float m1 = cml[qslot][0][lane][r];
      float l1 = cml[qslot][1][lane][r];
      float M  = fmaxf(mrow[r], m1);
      a0[r] = __builtin_amdgcn_exp2f(mrow[r] - M);
      a1[r] = __builtin_amdgcn_exp2f(m1 - M);
      inv[r] = 1.0f / (lrow[r] * a0[r] + l1 * a1[r]);
    }
    float* ob = out + (size_t)b * C_IN * N_TOK;
#pragma unroll
    for (int f = 0; f < 8; ++f) {
      v8f o1 = *(const v8f*)&clds[qslot][f][lane][0];
      const int ch = f * 16 + lo;
#pragma unroll
      for (int r = 0; r < 8; ++r)
        ob[(size_t)ch * N_TOK + qbase + r + 8 * hi] =
            (o[f][r] * a0[r] + o1[r] * a1[r]) * inv[r];
    }
  }
}

// ---------------------------------------------------------------------------
extern "C" void kernel_launch(void* const* d_in, const int* in_sizes, int n_in,
                              void* d_out, int out_size, void* d_ws, size_t ws_size,
                              hipStream_t stream) {
  const float* x  = (const float*)d_in[0];
  const float* y  = (const float*)d_in[1];
  const float* wq = (const float*)d_in[2];
  const float* wk = (const float*)d_in[3];
  const float* wv = (const float*)d_in[4];
  float* out = (float*)d_out;

  _Float16* qf  = (_Float16*)d_ws;                       //  2*9216*16  f16
  _Float16* ktf = qf  + (size_t)2 * N_TOK * CQ;          //  2*9216*16  f16
  _Float16* vfp = ktf + (size_t)2 * N_TOK * CQ;          //  2*128*9216 f16  (~5.9 MB total)

  qk_proj_kernel<<<dim3(N_TOK / 256, 2), 256, 0, stream>>>(y, wq, wk, qf, ktf);
  v_proj_kernel <<<dim3(N_TOK / 256, C_IN, 2), 256, 0, stream>>>(x, wv, vfp);
  flash_attn_kernel<<<dim3(N_TOK / 64, 2), 256, 0, stream>>>(qf, ktf, vfp, out);
}
